// MSMN_2224793059784
// MI455X (gfx1250) — compile-verified, matrix operand order checked by reading, not run on hardware
//
#include <hip/hip_runtime.h>
#include <hip/hip_bf16.h>

// ---------------------------------------------------------------------------
// Problem constants (from reference)
// ---------------------------------------------------------------------------
#define HDIM 768         // hidden size
#define KCLS 50          // synonym classes
#define SDIM 8           // synonyms per class
#define BDIM 16          // batch
#define LDIM 4096        // sequence length
#define QDIM (KCLS*SDIM) // 400 queries (25 tiles of 16)

// Workspace layout (bytes)
#define KEYS_BF_BYTES ((size_t)BDIM*LDIM*HDIM*2)   // 100,663,296
#define NOTE_T_BYTES  ((size_t)BDIM*LDIM*HDIM*2)   // 100,663,296 (transposed [B,H,L])
#define QBF_BYTES     ((size_t)QDIM*HDIM*2)        //     614,400
#define WKBF_BYTES    ((size_t)HDIM*HDIM*2)        //   1,179,648
#define CTX_BYTES     ((size_t)BDIM*QDIM*HDIM*4)   //  19,660,800

#define KEYS_BF_OFF 0
#define NOTE_T_OFF  (KEYS_BF_OFF + KEYS_BF_BYTES)
#define QBF_OFF     (NOTE_T_OFF + NOTE_T_BYTES)
#define WKBF_OFF    (QBF_OFF + QBF_BYTES)
#define CTX_OFF     (WKBF_OFF + WKBF_BYTES)

// CDNA5 async global->LDS path (probe-confirmed present on this toolchain)
#if defined(__gfx1250__) && \
    __has_builtin(__builtin_amdgcn_global_load_async_to_lds_b128) && \
    __has_builtin(__builtin_amdgcn_s_wait_asynccnt)
#define USE_ASYNC_LDS 1
#else
#define USE_ASYNC_LDS 0
#endif

// ---------------------------------------------------------------------------
// WMMA types / helpers (CDNA5 wave32, v_wmma_f32_16x16x32_bf16)
// ---------------------------------------------------------------------------
typedef __attribute__((ext_vector_type(16))) __bf16     v16bf;
typedef __attribute__((ext_vector_type(8)))  float      v8f;
typedef __attribute__((ext_vector_type(8)))  unsigned   v8u;
typedef __attribute__((ext_vector_type(4)))  int        v4i;

#if USE_ASYNC_LDS
typedef __attribute__((address_space(1))) v4i* gas_b128_t;  // global 128b payload
typedef __attribute__((address_space(3))) v4i* las_b128_t;  // LDS 128b payload
#endif

__device__ __forceinline__ v8f wmma_bf16(v8u a, v8u b, v8f c) {
  // D = A(16x32 bf16) * B(32x16 bf16) + C(16x16 f32)
  return __builtin_amdgcn_wmma_f32_16x16x32_bf16(
      false, __builtin_bit_cast(v16bf, a),
      false, __builtin_bit_cast(v16bf, b),
      (short)0, c, false, false);
}

// Hot-path conversion: TRUNCATE f32->bf16 pair -> one dword.
// (hi & 0xFFFF0000) | (lo >> 16) is a single v_perm_b32-class op; keeps the
// WMMA inner loops free of the 6-VALU RNE chains (this toolchain has no
// v_cvt_pk_bf16_f32 builtin). <=1 ulp bias; accumulation stays f32.
__device__ __forceinline__ unsigned pack_bf16_trunc(float lo, float hi) {
  return (__float_as_uint(hi) & 0xFFFF0000u) | (__float_as_uint(lo) >> 16);
}

// Offline-path conversion: round-to-nearest-even (preprocessing kernels only).
__device__ __forceinline__ unsigned short f32_to_bf16(float f) {
  unsigned u = __float_as_uint(f);
  u += 0x7FFFu + ((u >> 16) & 1u);
  return (unsigned short)(u >> 16);
}

__device__ __forceinline__ unsigned ld32u(const unsigned short* p) {
  return *(const unsigned*)p;        // two packed bf16, little-endian: low = k
}

// A-fragment (16x32 bf16) from row-major bf16 [rows, ld]; also the B-fragment
// for A @ Bt^T with Bt row-major [N, K] (N per lane mirrors M per lane).
// ISA 7.12.2: lane<16 -> row M=lane, K {half*8..+7, 16+half*8..+7}, dword-packed.
__device__ __forceinline__ v8u load_frag_rm(const unsigned short* p0, int ld, int lane) {
  const int half = lane >> 4;
  const unsigned short* p = p0 + (size_t)(lane & 15) * ld + half * 8;
  v8u f;
  f[0] = ld32u(p + 0);  f[1] = ld32u(p + 2);
  f[2] = ld32u(p + 4);  f[3] = ld32u(p + 6);
  f[4] = ld32u(p + 16); f[5] = ld32u(p + 18);
  f[6] = ld32u(p + 20); f[7] = ld32u(p + 22);
  return f;
}

// Same fragment but sourced from row-major fp32 (truncating pack, hot loop).
__device__ __forceinline__ v8u load_frag_rm_f32(const float* p0, int ld, int lane) {
  const int half = lane >> 4;
  const float* p = p0 + (size_t)(lane & 15) * ld + half * 8;
  v8u f;
#pragma unroll
  for (int v = 0; v < 4; ++v) {
    f[v]     = pack_bf16_trunc(p[2*v],      p[2*v + 1]);
    f[4 + v] = pack_bf16_trunc(p[16 + 2*v], p[16 + 2*v + 1]);
  }
  return f;
}

// ---------------------------------------------------------------------------
// Kernel 1: fp32 -> bf16 conversion (grid-stride; used for Wk only)
// ---------------------------------------------------------------------------
__global__ void cvt_bf16_kernel(const float* __restrict__ src,
                                unsigned short* __restrict__ dst, long n) {
  long i = (long)blockIdx.x * blockDim.x + threadIdx.x;
  long stride = (long)gridDim.x * blockDim.x;
  for (; i < n; i += stride) dst[i] = f32_to_bf16(src[i]);
}

// ---------------------------------------------------------------------------
// Kernel 2: tiled transpose note_h [B,L,H] f32 -> note_t [B,H,L] bf16
// 32x32 tiles, coalesced on both sides, LDS staging with conflict padding.
// ---------------------------------------------------------------------------
__global__ __launch_bounds__(256) void transpose_bf16_kernel(
    const float* __restrict__ src, unsigned short* __restrict__ dst) {
  __shared__ float tile[32][33];
  const int b  = blockIdx.z;
  const int l0 = blockIdx.x * 32;
  const int h0 = blockIdx.y * 32;
  const int tx = threadIdx.x;   // 0..31
  const int ty = threadIdx.y;   // 0..7
  const float* s = src + ((size_t)b * LDIM + l0) * HDIM + h0;
#pragma unroll
  for (int i = 0; i < 4; ++i)
    tile[ty + 8 * i][tx] = s[(size_t)(ty + 8 * i) * HDIM + tx];
  __syncthreads();
  unsigned short* d = dst + ((size_t)b * HDIM + h0) * LDIM + l0;
#pragma unroll
  for (int i = 0; i < 4; ++i)
    d[(size_t)(ty + 8 * i) * LDIM + tx] = f32_to_bf16(tile[tx][ty + 8 * i]);
}

// ---------------------------------------------------------------------------
// Kernel 3: Q = syn_cls @ Wq^T + bq  -> bf16 [400, 768]   (tiny: scalar path)
// ---------------------------------------------------------------------------
__global__ __launch_bounds__(256) void qproj_kernel(
    const float* __restrict__ cls, const float* __restrict__ Wq,
    const float* __restrict__ bq, unsigned short* __restrict__ q_bf) {
  int idx = blockIdx.x * 256 + threadIdx.x;
  if (idx >= QDIM * HDIM) return;
  int r = idx / HDIM, c = idx % HDIM;
  const float* a = cls + (size_t)r * HDIM;
  const float* w = Wq + (size_t)c * HDIM;
  float s = bq[c];
  for (int h = 0; h < HDIM; ++h) s = fmaf(a[h], w[h], s);
  q_bf[idx] = f32_to_bf16(s);
}

// ---------------------------------------------------------------------------
// Kernel 4: Keys = note @ Wk^T + bk -> bf16 [B*L, 768]  (WMMA GEMM)
// A read directly from fp32 note_h (truncating in-register pack).
// block = 8 waves; wave w: rows m0 = bx*128 + 16w, cols n0 = by*64 (4 N-tiles)
// ---------------------------------------------------------------------------
__global__ __launch_bounds__(256) void keys_gemm_kernel(
    const float* __restrict__ note_f32,
    const unsigned short* __restrict__ wk_bf,
    const float* __restrict__ bk,
    unsigned short* __restrict__ keys_bf) {
  const int w    = threadIdx.x >> 5;
  const int lane = threadIdx.x & 31;
  const int half = lane >> 4;
  const int lr   = lane & 15;
  const int m0 = blockIdx.x * 128 + w * 16;
  const int n0 = blockIdx.y * 64;

  v8f acc[4];
#pragma unroll
  for (int t = 0; t < 4; ++t) acc[t] = (v8f){};

  const float* aptr = note_f32 + (size_t)m0 * HDIM;
#pragma unroll 2
  for (int k = 0; k < HDIM; k += 32) {
    v8u b0 = load_frag_rm(wk_bf + (size_t)(n0 +  0) * HDIM + k, HDIM, lane);
    v8u b1 = load_frag_rm(wk_bf + (size_t)(n0 + 16) * HDIM + k, HDIM, lane);
    v8u b2 = load_frag_rm(wk_bf + (size_t)(n0 + 32) * HDIM + k, HDIM, lane);
    v8u b3 = load_frag_rm(wk_bf + (size_t)(n0 + 48) * HDIM + k, HDIM, lane);
    v8u a  = load_frag_rm_f32(aptr + k, HDIM, lane);
    acc[0] = wmma_bf16(a, b0, acc[0]);
    acc[1] = wmma_bf16(a, b1, acc[1]);
    acc[2] = wmma_bf16(a, b2, acc[2]);
    acc[3] = wmma_bf16(a, b3, acc[3]);
  }
  // D layout: lane<16 -> (M=r, N=lane); lane>=16 -> (M=8+r, N=lane-16)
#pragma unroll
  for (int t = 0; t < 4; ++t) {
    int col = n0 + t * 16 + lr;
    float bias = bk[col];
#pragma unroll
    for (int r = 0; r < 8; ++r) {
      int row = m0 + r + 8 * half;
      keys_bf[(size_t)row * HDIM + col] = f32_to_bf16(acc[t][r] + bias);
    }
  }
}

// ---------------------------------------------------------------------------
// Kernel 5: fused attention (flash-style, per (batch, 16-query tile))
//   S = Q Kt^T * scale + mask ; online softmax over L ; O += P @ V
// Q tile staged in LDS once (async global->LDS). 8 waves: each computes one
// 16x16 logit sub-tile per 128-key chunk and owns a 96-wide H slice of the O
// accumulator. V fragments come from the transposed note_t[B,H,L] so every
// load is a contiguous b128.
// ---------------------------------------------------------------------------
__global__ __launch_bounds__(256) void attn_kernel(
    const unsigned short* __restrict__ q_bf,     // [400, 768] bf16
    const unsigned short* __restrict__ keys_all, // [B, L, 768] bf16
    const unsigned short* __restrict__ note_t,   // [B, 768, L] bf16 (transposed)
    const int* __restrict__ amask,               // [B, L]
    float* __restrict__ ctx) {                   // [B, 400, 768] f32
  __shared__ unsigned short Qsm[16 * HDIM];      // 24 KB
  __shared__ float          Ssm[16][128];        //  8 KB
  __shared__ unsigned short Psm[16][128];        //  4 KB
  __shared__ float mrun[16], lrun[16], alpha[16];

  const int b    = blockIdx.y;
  const int q0   = blockIdx.x * 16;
  const int w    = threadIdx.x >> 5;
  const int lane = threadIdx.x & 31;
  const int half = lane >> 4;
  const int lr   = lane & 15;
  const int h0   = w * 96;
  const float scale = 0.036084391826f;  // 768^-0.5

  const unsigned short* keys = keys_all + (size_t)b * LDIM * HDIM;
  const unsigned short* vmat = note_t + (size_t)b * HDIM * LDIM;

  // ---- stage Q tile [16 x 768] bf16 into LDS (async where available) ------
  {
    const unsigned short* qsrc = q_bf + (size_t)q0 * HDIM;
#if USE_ASYNC_LDS
#pragma unroll
    for (int i = 0; i < 6; ++i) {                 // 6 * 256 threads * 16B = 24 KB
      int elt = (i * 256 + threadIdx.x) * 8;      // 8 ushorts = 16 bytes per thread
      __builtin_amdgcn_global_load_async_to_lds_b128(
          (gas_b128_t)(void*)(qsrc + elt),
          (las_b128_t)(void*)(&Qsm[elt]), 0, 0);
    }
    __builtin_amdgcn_s_wait_asynccnt(0);
#else
#pragma unroll
    for (int i = 0; i < 6; ++i) {
      int elt = (i * 256 + threadIdx.x) * 8;
      *(uint4*)(&Qsm[elt]) = *(const uint4*)(qsrc + elt);
    }
#endif
  }
  if (threadIdx.x < 16) { mrun[threadIdx.x] = -3.0e38f; lrun[threadIdx.x] = 0.0f; }

  v8f acc[6];
#pragma unroll
  for (int t = 0; t < 6; ++t) acc[t] = (v8f){};
  __syncthreads();

  for (int l0 = 0; l0 < LDIM; l0 += 128) {
    // -------- Phase 1: logits S[16q x 16l] per wave ----------------------
    const int lsub = l0 + w * 16;
    if (l0 + 128 < LDIM)   // prefetch next chunk's key rows (global_prefetch_b8)
      __builtin_prefetch(keys + (size_t)(lsub + 128) * HDIM, 0, 0);
    v8f s = (v8f){};
#pragma unroll 4
    for (int k = 0; k < HDIM; k += 32) {
      v8u kb = load_frag_rm(keys + (size_t)lsub * HDIM + k, HDIM, lane);
      v8u a  = load_frag_rm(Qsm + k, HDIM, lane);                       // ds path
      s = wmma_bf16(a, kb, s);
    }
    const float madd =
        (1.0f - (float)amask[b * LDIM + lsub + lr]) * -1.0e9f;
#pragma unroll
    for (int r = 0; r < 8; ++r)
      Ssm[r + 8 * half][w * 16 + lr] = s[r] * scale + madd;
    __syncthreads();

    // -------- Phase 2: online softmax, all 8 waves (2 rows each) ---------
    {
      const int row = 2 * w + half;      // 16 rows over 8 waves
      const int c0  = lr * 8;            // 8 columns per lane
      float m_old = mrun[row];
      float cmax = -3.0e38f;
#pragma unroll
      for (int j = 0; j < 8; ++j) cmax = fmaxf(cmax, Ssm[row][c0 + j]);
#pragma unroll
      for (int off = 8; off > 0; off >>= 1)
        cmax = fmaxf(cmax, __shfl_xor(cmax, off, 16));
      const float m_new = fmaxf(m_old, cmax);
      const float al = __expf(m_old - m_new);
      float sum = 0.0f;
#pragma unroll
      for (int j = 0; j < 4; ++j) {      // dword-packed P stores (trunc pack)
        float p0 = __expf(Ssm[row][c0 + 2*j]     - m_new);
        float p1 = __expf(Ssm[row][c0 + 2*j + 1] - m_new);
        *(unsigned*)&Psm[row][c0 + 2*j] = pack_bf16_trunc(p0, p1);
        sum += p0 + p1;
      }
#pragma unroll
      for (int off = 8; off > 0; off >>= 1)
        sum += __shfl_xor(sum, off, 16);
      if (lr == 0) {
        lrun[row] = lrun[row] * al + sum;
        mrun[row] = m_new;
        alpha[row] = al;
      }
    }
    __syncthreads();

    // -------- Phase 3: rescale accumulators, O += P @ V ------------------
#pragma unroll
    for (int r = 0; r < 8; ++r) {
      float al = alpha[r + 8 * half];
#pragma unroll
      for (int t = 0; t < 6; ++t) acc[t][r] *= al;
    }
#pragma unroll 2
    for (int kk = 0; kk < 128; kk += 32) {
      // load all 6 V fragments, then the WMMA group (overlap loads/WMMA)
      v8u vb[6];
#pragma unroll
      for (int t = 0; t < 6; ++t)
        vb[t] = load_frag_rm(vmat + (size_t)(h0 + t * 16) * LDIM + l0 + kk,
                             LDIM, lane);
      v8u a = load_frag_rm(&Psm[0][0] + kk, 128, lane);   // ds path
#pragma unroll
      for (int t = 0; t < 6; ++t) acc[t] = wmma_bf16(a, vb[t], acc[t]);
    }
    // Next phase-1 writes Ssm only; Psm/alpha are rewritten after the next
    // barrier, so two barriers per iteration are sufficient.
  }

  // -------- epilogue: O / l, store fp32 context --------------------------
#pragma unroll
  for (int r = 0; r < 8; ++r) {
    const int qrow = r + 8 * half;
    const float inv = 1.0f / lrun[qrow];
#pragma unroll
    for (int t = 0; t < 6; ++t) {
      ctx[(((size_t)b * QDIM) + q0 + qrow) * HDIM + h0 + t * 16 + lr] =
          acc[t][r] * inv;
    }
  }
}

// ---------------------------------------------------------------------------
// Kernel 6: logits[b,k] = sum_h ( max_s ctx[b,k*8+s,h] ) * Wc[h] + bc
// ---------------------------------------------------------------------------
__global__ __launch_bounds__(256) void finalize_kernel(
    const float* __restrict__ ctx, const float* __restrict__ Wc,
    const float* __restrict__ bc, float* __restrict__ out) {
  __shared__ float red[256];
  const int bk = blockIdx.x;           // 0 .. B*KCLS-1
  const int b = bk / KCLS, k = bk % KCLS;
  float part = 0.0f;
  for (int h = threadIdx.x; h < HDIM; h += 256) {
    const float* base = ctx + (((size_t)b * QDIM) + k * SDIM) * HDIM + h;
    float mx = base[0];
#pragma unroll
    for (int s = 1; s < SDIM; ++s) mx = fmaxf(mx, base[(size_t)s * HDIM]);
    part += mx * Wc[h];
  }
  red[threadIdx.x] = part;
  __syncthreads();
  for (int off = 128; off > 0; off >>= 1) {
    if (threadIdx.x < off) red[threadIdx.x] += red[threadIdx.x + off];
    __syncthreads();
  }
  if (threadIdx.x == 0) out[bk] = red[0] + bc[0];
}

// ---------------------------------------------------------------------------
// Host entry
// ---------------------------------------------------------------------------
extern "C" void kernel_launch(void* const* d_in, const int* in_sizes, int n_in,
                              void* d_out, int out_size, void* d_ws, size_t ws_size,
                              hipStream_t stream) {
  const float* note_h = (const float*)d_in[0];   // [B, L, H]
  const int*   amask  = (const int*)d_in[1];     // [B, L]
  const float* syn    = (const float*)d_in[2];   // [K, S, H]
  const float* Wq     = (const float*)d_in[3];   // [H, H]
  const float* bq     = (const float*)d_in[4];   // [H]
  const float* Wk     = (const float*)d_in[5];   // [H, H]
  const float* bk     = (const float*)d_in[6];   // [H]
  const float* Wc     = (const float*)d_in[7];   // [1, H]
  const float* bc     = (const float*)d_in[8];   // [1]
  float* out = (float*)d_out;

  char* ws = (char*)d_ws;
  unsigned short* keys_bf = (unsigned short*)(ws + KEYS_BF_OFF);
  unsigned short* note_t  = (unsigned short*)(ws + NOTE_T_OFF);
  unsigned short* q_bf    = (unsigned short*)(ws + QBF_OFF);
  unsigned short* wk_bf   = (unsigned short*)(ws + WKBF_OFF);
  float*          ctx     = (float*)(ws + CTX_OFF);

  // 1) Wk -> bf16
  cvt_bf16_kernel<<<256, 256, 0, stream>>>(Wk, wk_bf, (long)HDIM * HDIM);

  // 2) note_h [B,L,H] f32 -> note_t [B,H,L] bf16 (for contiguous V fragments)
  dim3 tgrid(LDIM / 32, HDIM / 32, BDIM);
  transpose_bf16_kernel<<<tgrid, dim3(32, 8), 0, stream>>>(note_h, note_t);

  // 3) Q projection (tiny)
  qproj_kernel<<<(QDIM * HDIM + 255) / 256, 256, 0, stream>>>(syn, Wq, bq, q_bf);

  // 4) Keys GEMM: [B*L, H] x [H, H]^T  (WMMA, fp32 A packed in-register)
  dim3 kgrid(BDIM * LDIM / 128, HDIM / 64);
  keys_gemm_kernel<<<kgrid, 256, 0, stream>>>(note_h, wk_bf, bk, keys_bf);

  // 5) fused attention (WMMA + online softmax + async Q staging)
  dim3 agrid(QDIM / 16, BDIM);
  attn_kernel<<<agrid, 256, 0, stream>>>(q_bf, keys_bf, note_t, amask, ctx);

  // 6) max-over-synonyms + classifier dot
  finalize_kernel<<<BDIM * KCLS, 256, 0, stream>>>(ctx, Wc, bc, out);
}
